// RBACGraphAttentionNetwork_10136122819016
// MI455X (gfx1250) — compile-verified, matrix operand order checked by reading, not run on hardware
//
#include <hip/hip_runtime.h>
#include <hip/hip_bf16.h>
#include <math.h>

typedef __attribute__((ext_vector_type(16))) __bf16 v16bf;
typedef __attribute__((ext_vector_type(8)))  float  v8f;

// ---------- helpers ----------

__device__ __forceinline__ __bf16 f2bf(float f) {
    return static_cast<__bf16>(f);   // native v_cvt (RNE) on gfx1250
}

__device__ __forceinline__ void atomicMaxFloat(float* addr, float v) {
    // works for mixed-sign values given init = -inf
    if (v >= 0.0f) atomicMax((int*)addr, __float_as_int(v));
    else           atomicMin((unsigned int*)addr, (unsigned int)__float_as_int(v));
}

// ---------- weight pre-swizzle: fp32 [K x 128] -> bf16 B-fragment order ----------
// B fragment layout (32x16 tile, K x N): lane = nn + 16*((kk%16)/8),
// elem = (kk%8) + 8*(kk/16). Tile (kt,nt) at base ((kt*8+nt)*32 + lane)*16.
__global__ void convw_kernel(const float* __restrict__ W, __bf16* __restrict__ Wb, int K) {
    int t = blockIdx.x * blockDim.x + threadIdx.x;
    int total = K * 128;
    if (t >= total) return;
    int k = t >> 7, n = t & 127;
    int kt = k >> 5, nt = n >> 4, kk = k & 31, nn = n & 15;
    int lane = nn + (((kk & 15) >> 3) << 4);
    int elem = (kk & 7) + ((kk >> 4) << 3);
    Wb[((((kt << 3) + nt) * 32 + lane) << 4) + elem] = f2bf(W[t]);
}

// ---------- WMMA GEMM: C[M x 128] = A[M x K] (fp32, row-major) x Wb (bf16 swizzled) ----------
// 8 waves / block, each wave owns a 16-row strip, 8 N-tiles wide, K stepped by 32.
// B (<= 64 KB) is staged once per block into LDS: all 8 waves reuse it via ds_load_b128.
__global__ void gemm_wmma_kernel(const float* __restrict__ A, const __bf16* __restrict__ Bw,
                                 float* __restrict__ C, int M, int K) {
    __shared__ __align__(32) __bf16 Bs[256 * 128];     // sized for max K=256 (64 KB)

    // cooperative global -> LDS copy of the swizzled B panel (K*128 bf16 = K*256 bytes)
    {
        const uint4* src = (const uint4*)Bw;
        uint4* dst = (uint4*)Bs;
        int nvec = (K * 128) >> 3;                      // 8 bf16 per uint4
        for (int i = threadIdx.x; i < nvec; i += blockDim.x) dst[i] = src[i];
    }
    __syncthreads();

    int wave = threadIdx.x >> 5;
    int lane = threadIdx.x & 31;
    int rowBase = blockIdx.x * 128 + wave * 16;
    if (rowBase >= M) return;                           // wave-uniform: EXEC stays full

    v8f acc[8];
#pragma unroll
    for (int i = 0; i < 8; i++)
#pragma unroll
        for (int j = 0; j < 8; j++) acc[i][j] = 0.0f;

    int m  = lane & 15;
    int kb = (lane >> 4) << 3;                          // 0 or 8
    const float* arow = A + (size_t)(rowBase + m) * K;
    const v16bf* bfrag_base = (const v16bf*)Bs;         // frag idx = (kt*8+nt)*32 + lane
    int ksteps = K >> 5;

    for (int kt = 0; kt < ksteps; kt++) {
        const float* ap = arow + (kt << 5) + kb;
        if (kt + 1 < ksteps) __builtin_prefetch(ap + 32, 0, 1);  // global_prefetch next A tile
        v16bf af;
#pragma unroll
        for (int j = 0; j < 8; j++) {                   // ISA 16-bit A layout (16x32)
            af[j]     = f2bf(ap[j]);                    // K = kb + j
            af[8 + j] = f2bf(ap[16 + j]);               // K = kb + 16 + j
        }
        // batch the 8 LDS fragment reads, then the 8 WMMAs, to overlap ds latency
        v16bf bfr[8];
#pragma unroll
        for (int nt = 0; nt < 8; nt++)
            bfr[nt] = bfrag_base[((kt << 3) + nt) * 32 + lane];
#pragma unroll
        for (int nt = 0; nt < 8; nt++)
            acc[nt] = __builtin_amdgcn_wmma_f32_16x16x32_bf16(
                false, af, false, bfr[nt], (short)0, acc[nt], false, false);
    }

    int nn = lane & 15;
    int mb = (lane >> 4) << 3;                          // C/D layout: vgpr i -> M = mb + i
#pragma unroll
    for (int nt = 0; nt < 8; nt++)
#pragma unroll
        for (int i = 0; i < 8; i++)
            C[(size_t)(rowBase + mb + i) * 128 + nt * 16 + nn] = acc[nt][i];
}

// ---------- per-(node,head): attention dots + re-init max/den + zero accumulator ----------
__global__ void al_init_kernel(const float* __restrict__ h, const float* __restrict__ a_src,
                               const float* __restrict__ a_dst, float* __restrict__ alS,
                               float* __restrict__ alD, float* __restrict__ mx,
                               float* __restrict__ den, float* __restrict__ acc, int N) {
    int t = blockIdx.x * blockDim.x + threadIdx.x;
    if (t >= N * 8) return;
    int n = t >> 3, hh = t & 7;
    const float* hp = h + (size_t)n * 128 + hh * 16;
    float s0 = 0.0f, s1 = 0.0f;
#pragma unroll
    for (int c = 0; c < 16; c++) {
        float v = hp[c];
        s0 += v * a_src[hh * 16 + c];
        s1 += v * a_dst[hh * 16 + c];
    }
    alS[t] = s0; alD[t] = s1;
    mx[t] = -__builtin_inff();
    den[t] = 0.0f;
    float* ap = acc + (size_t)n * 128 + hh * 16;
#pragma unroll
    for (int c = 0; c < 16; c++) ap[c] = 0.0f;
}

// ---------- per-(edge,head): segment max of leaky-relu logits ----------
__global__ void edge_max_kernel(const int* __restrict__ ei, const float* __restrict__ alS,
                                const float* __restrict__ alD, float* __restrict__ mx,
                                int N, int E) {
    int t = blockIdx.x * blockDim.x + threadIdx.x;
    if (t >= (E + N) * 8) return;
    int idx = t >> 3, hh = t & 7;
    int s, d;
    if (idx < E) { s = ei[idx]; d = ei[E + idx]; } else { s = d = idx - E; }  // self-loops
    float e = alS[s * 8 + hh] + alD[d * 8 + hh];
    e = e > 0.0f ? e : 0.2f * e;
    atomicMaxFloat(&mx[d * 8 + hh], e);
}

// ---------- per-(edge,head): exp weights, denom, weighted feature scatter ----------
__global__ void edge_sum_kernel(const int* __restrict__ ei, const float* __restrict__ alS,
                                const float* __restrict__ alD, const float* __restrict__ mx,
                                const float* __restrict__ h, float* __restrict__ den,
                                float* __restrict__ acc, int N, int E) {
    int t = blockIdx.x * blockDim.x + threadIdx.x;
    if (t >= (E + N) * 8) return;
    int idx = t >> 3, hh = t & 7;
    int s, d;
    if (idx < E) { s = ei[idx]; d = ei[E + idx]; } else { s = d = idx - E; }
    float e = alS[s * 8 + hh] + alD[d * 8 + hh];
    e = e > 0.0f ? e : 0.2f * e;
    float ex = __expf(e - mx[d * 8 + hh]);
    atomicAdd(&den[d * 8 + hh], ex);
    const float* hp = h + (size_t)s * 128 + hh * 16;
    float* ap = acc + (size_t)d * 128 + hh * 16;
#pragma unroll
    for (int c = 0; c < 16; c++) atomicAdd(&ap[c], hp[c] * ex);
}

// ---------- normalize + bias + ELU (layers 1,2; concat) ----------
__global__ void finalize_concat_kernel(float* __restrict__ acc, const float* __restrict__ den,
                                       const float* __restrict__ b, int N) {
    int t = blockIdx.x * blockDim.x + threadIdx.x;
    if (t >= N * 8) return;
    int n = t >> 3, hh = t & 7;
    float inv = 1.0f / den[t];
    float* ap = acc + (size_t)n * 128 + hh * 16;
#pragma unroll
    for (int c = 0; c < 16; c++) {
        float v = ap[c] * inv + b[hh * 16 + c];
        ap[c] = v > 0.0f ? v : (__expf(v) - 1.0f);   // ELU(alpha=1)
    }
}

// ---------- normalize + head-mean + bias (layer 3) -> d_out h[N,16] ----------
__global__ void finalize_mean_kernel(const float* __restrict__ acc, const float* __restrict__ den,
                                     const float* __restrict__ b3, float* __restrict__ out_h, int N) {
    int n = blockIdx.x * blockDim.x + threadIdx.x;
    if (n >= N) return;
    float invd[8];
#pragma unroll
    for (int hh = 0; hh < 8; hh++) invd[hh] = 1.0f / den[n * 8 + hh];
#pragma unroll
    for (int c = 0; c < 16; c++) {
        float s = 0.0f;
#pragma unroll
        for (int hh = 0; hh < 8; hh++) s += acc[(size_t)n * 128 + hh * 16 + c] * invd[hh];
        out_h[(size_t)n * 16 + c] = s * 0.125f + b3[c];
    }
}

__global__ void zero18_kernel(float* __restrict__ p) {
    if (threadIdx.x < 18) p[threadIdx.x] = 0.0f;
}

// ---------- anomaly MLP + graph-mean (LDS reduction -> global atomics) ----------
__global__ void heads_kernel(const float* __restrict__ h, const float* __restrict__ aw1,
                             const float* __restrict__ ab1, const float* __restrict__ aw2,
                             const float* __restrict__ ab2, float* __restrict__ anom,
                             float* __restrict__ emb, int N) {
    __shared__ float semb[16];
    if (threadIdx.x < 16) semb[threadIdx.x] = 0.0f;
    __syncthreads();
    int n = blockIdx.x * blockDim.x + threadIdx.x;
    if (n < N) {
        float hv[16];
#pragma unroll
        for (int c = 0; c < 16; c++) hv[c] = h[(size_t)n * 16 + c];
        float o = ab2[0];
        for (int j = 0; j < 32; j++) {
            float a = ab1[j];
#pragma unroll
            for (int c = 0; c < 16; c++) a += hv[c] * aw1[c * 32 + j];
            o += fmaxf(a, 0.0f) * aw2[j];
        }
        anom[n] = 1.0f / (1.0f + __expf(-o));
        float invN = 1.0f / (float)N;
#pragma unroll
        for (int c = 0; c < 16; c++) atomicAdd(&semb[c], hv[c] * invN);  // ds_add_f32
    }
    __syncthreads();
    if (threadIdx.x < 16) atomicAdd(&emb[threadIdx.x], semb[threadIdx.x]);
}

// ---------- graph classifier 16 -> 64 -> 2 ----------
__global__ void pred_kernel(const float* __restrict__ emb, const float* __restrict__ w1,
                            const float* __restrict__ b1, const float* __restrict__ w2,
                            const float* __restrict__ b2, float* __restrict__ pred) {
    __shared__ float hid[64];
    int j = threadIdx.x;
    float a = b1[j];
#pragma unroll
    for (int c = 0; c < 16; c++) a += emb[c] * w1[c * 64 + j];
    hid[j] = fmaxf(a, 0.0f);
    __syncthreads();
    if (j < 2) {
        float o = b2[j];
#pragma unroll
        for (int k = 0; k < 64; k++) o += hid[k] * w2[k * 2 + j];
        pred[j] = o;
    }
}

extern "C" void kernel_launch(void* const* d_in, const int* in_sizes, int n_in,
                              void* d_out, int out_size, void* d_ws, size_t ws_size,
                              hipStream_t stream) {
    const float* x   = (const float*)d_in[0];
    const int*   ei  = (const int*)d_in[1];
    const float* W1  = (const float*)d_in[2];
    const float* a1s = (const float*)d_in[3];
    const float* a1d = (const float*)d_in[4];
    const float* b1  = (const float*)d_in[5];
    const float* W2  = (const float*)d_in[6];
    const float* a2s = (const float*)d_in[7];
    const float* a2d = (const float*)d_in[8];
    const float* b2  = (const float*)d_in[9];
    const float* W3  = (const float*)d_in[10];
    const float* a3s = (const float*)d_in[11];
    const float* a3d = (const float*)d_in[12];
    const float* b3  = (const float*)d_in[13];
    const float* cw1 = (const float*)d_in[14];
    const float* cb1 = (const float*)d_in[15];
    const float* cw2 = (const float*)d_in[16];
    const float* cb2 = (const float*)d_in[17];
    const float* aw1 = (const float*)d_in[18];
    const float* ab1 = (const float*)d_in[19];
    const float* aw2 = (const float*)d_in[20];
    const float* ab2 = (const float*)d_in[21];

    const int N = in_sizes[0] / 256;   // 50000
    const int E = in_sizes[1] / 2;     // 800000

    float*  buf0 = (float*)d_ws;                       // [N,128] gemm output
    float*  buf1 = buf0 + (size_t)N * 128;             // [N,128] aggregation / next input
    float*  alS  = buf1 + (size_t)N * 128;             // [N,8]
    float*  alD  = alS + (size_t)N * 8;
    float*  mx   = alD + (size_t)N * 8;
    float*  den  = mx  + (size_t)N * 8;
    __bf16* Wb   = (__bf16*)(den + (size_t)N * 8);     // 256*128 bf16 swizzled weights

    float* out_h    = (float*)d_out;                   // [N,16]
    float* out_an   = out_h + (size_t)N * 16;          // [N]
    float* out_emb  = out_an + N;                      // [16]
    float* out_pred = out_emb + 16;                    // [2]

    const int tpb = 256;
    const int gNH = (N * 8 + tpb - 1) / tpb;
    const int gEH = ((E + N) * 8 + tpb - 1) / tpb;
    const int gM  = (N + 127) / 128;
    const int gN  = (N + tpb - 1) / tpb;

    // ----- layer 1: 256 -> 8x16 concat, ELU -----
    convw_kernel<<<(256 * 128 + tpb - 1) / tpb, tpb, 0, stream>>>(W1, Wb, 256);
    gemm_wmma_kernel<<<gM, tpb, 0, stream>>>(x, Wb, buf0, N, 256);
    al_init_kernel<<<gNH, tpb, 0, stream>>>(buf0, a1s, a1d, alS, alD, mx, den, buf1, N);
    edge_max_kernel<<<gEH, tpb, 0, stream>>>(ei, alS, alD, mx, N, E);
    edge_sum_kernel<<<gEH, tpb, 0, stream>>>(ei, alS, alD, mx, buf0, den, buf1, N, E);
    finalize_concat_kernel<<<gNH, tpb, 0, stream>>>(buf1, den, b1, N);

    // ----- layer 2: 128 -> 8x16 concat, ELU -----
    convw_kernel<<<(128 * 128 + tpb - 1) / tpb, tpb, 0, stream>>>(W2, Wb, 128);
    gemm_wmma_kernel<<<gM, tpb, 0, stream>>>(buf1, Wb, buf0, N, 128);
    al_init_kernel<<<gNH, tpb, 0, stream>>>(buf0, a2s, a2d, alS, alD, mx, den, buf1, N);
    edge_max_kernel<<<gEH, tpb, 0, stream>>>(ei, alS, alD, mx, N, E);
    edge_sum_kernel<<<gEH, tpb, 0, stream>>>(ei, alS, alD, mx, buf0, den, buf1, N, E);
    finalize_concat_kernel<<<gNH, tpb, 0, stream>>>(buf1, den, b2, N);

    // ----- layer 3: 128 -> mean over heads -> [N,16] -----
    convw_kernel<<<(128 * 128 + tpb - 1) / tpb, tpb, 0, stream>>>(W3, Wb, 128);
    gemm_wmma_kernel<<<gM, tpb, 0, stream>>>(buf1, Wb, buf0, N, 128);
    al_init_kernel<<<gNH, tpb, 0, stream>>>(buf0, a3s, a3d, alS, alD, mx, den, buf1, N);
    edge_max_kernel<<<gEH, tpb, 0, stream>>>(ei, alS, alD, mx, N, E);
    edge_sum_kernel<<<gEH, tpb, 0, stream>>>(ei, alS, alD, mx, buf0, den, buf1, N, E);
    finalize_mean_kernel<<<gN, tpb, 0, stream>>>(buf1, den, b3, out_h, N);

    // ----- heads -----
    zero18_kernel<<<1, 32, 0, stream>>>(out_emb);
    heads_kernel<<<gN, tpb, 0, stream>>>(out_h, aw1, ab1, aw2, ab2, out_an, out_emb, N);
    pred_kernel<<<1, 64, 0, stream>>>(out_emb, cw1, cb1, cw2, cb2, out_pred);
}
